// CISTransformerDecoder_73504070303948
// MI455X (gfx1250) — compile-verified, matrix operand order checked by reading, not run on hardware
//
#include <hip/hip_runtime.h>
#include <hip/hip_bf16.h>
#include <cstdint>
#include <cstddef>

// ---------------- model constants ----------------
#define LQ      5440
#define CDIM    256
#define DFFN    1024
#define NHEADS  8
#define NLEV    4
#define NPTS    4
#define NLAYERS 6
#define DH      32          // CDIM / NHEADS == wave32: lane == channel

typedef __attribute__((ext_vector_type(16))) __bf16 v16bf;
typedef __attribute__((ext_vector_type(8)))  __bf16 v8bf;
typedef __attribute__((ext_vector_type(8)))  float  v8f;

static __device__ __forceinline__ __bf16 f2bf(float f) {
  union { float f; uint32_t u; } x; x.f = f;
  uint32_t r = x.u + 0x7FFFu + ((x.u >> 16) & 1u);   // round-to-nearest-even
  uint16_t h = (uint16_t)(r >> 16);
  return __builtin_bit_cast(__bf16, h);
}

// ---------------- weight transpose + cast: W[K][N] (per layer) -> Wt[N][K] bf16 ----------------
__global__ void k_wt_cast(const float* __restrict__ W, __bf16* __restrict__ Wt, int K, int N) {
  int layer = blockIdx.y;
  int idx = blockIdx.x * 256 + threadIdx.x;
  int kn = K * N;
  if (idx < kn) {
    int k = idx / N, n = idx % N;
    Wt[(size_t)layer * kn + (size_t)n * K + k] = f2bf(W[(size_t)layer * kn + idx]);
  }
}

// ---------------- per-level prep: concat + level embed + value_in ----------------
__global__ void k_prep_level(const float* __restrict__ tgt, const float* __restrict__ pos,
                             const float* __restrict__ mem, const float* __restrict__ posm,
                             const float* __restrict__ le_row,
                             float* __restrict__ pe, float* __restrict__ posraw,
                             float* __restrict__ out0,
                             float* __restrict__ vin, __bf16* __restrict__ vinbf,
                             int n, int qoff) {
  int idx = blockIdx.x;        // spatial index within level (y*w + x)
  int c = threadIdx.x;         // channel
  int q = qoff + idx;
  float pv = pos[(size_t)idx * CDIM + c];
  float le = le_row[c];
  pe[(size_t)q * CDIM + c] = pv + le;
  posraw[(size_t)q * CDIM + c] = pv;
  out0[(size_t)q * CDIM + c] = tgt[(size_t)idx * CDIM + c];
  // mem/posm are [C, h, w] -> value_in row-major [q, C]
  float v = mem[(size_t)c * n + idx] + posm[(size_t)c * n + idx] + le;
  vin[(size_t)q * CDIM + c] = v;
  vinbf[(size_t)q * CDIM + c] = f2bf(v);
}

// ---------------- reference points: sigmoid(pos @ W_ref + b_ref); wave per query ----------------
__global__ void k_ref(const float* __restrict__ posraw, const float* __restrict__ Wref,
                      const float* __restrict__ bref, float* __restrict__ ref) {
  int q = blockIdx.x * 8 + (threadIdx.x >> 5);
  int lane = threadIdx.x & 31;
  float s0 = 0.f, s1 = 0.f;
#pragma unroll
  for (int i = 0; i < 8; ++i) {
    int c = lane + i * 32;
    float p = posraw[(size_t)q * CDIM + c];
    s0 += p * Wref[c * 2 + 0];
    s1 += p * Wref[c * 2 + 1];
  }
#pragma unroll
  for (int m = 16; m >= 1; m >>= 1) { s0 += __shfl_xor(s0, m, 32); s1 += __shfl_xor(s1, m, 32); }
  if (lane == 0) {
    ref[q * 2 + 0] = 1.f / (1.f + expf(-(s0 + bref[0])));
    ref[q * 2 + 1] = 1.f / (1.f + expf(-(s1 + bref[1])));
  }
}

// ---------------- q = out + pos_embed, cast bf16 ----------------
__global__ void k_qcast(const float* __restrict__ a, const float* __restrict__ b,
                        __bf16* __restrict__ o) {
  size_t i = (size_t)blockIdx.x * 256 + threadIdx.x;
  o[i] = f2bf(a[i] + b[i]);
}

// ---------------- bf16 WMMA GEMM: D[M,N] = A[M,K] * Bt[N,K]^T + bias ----------------
// block = 256 threads = 8 waves; each wave computes NSUB adjacent 16x16 tiles along N,
// reusing the A fragment NSUB times (register blocking). grid = (M/16, N/(128*NSUB)).
template <int NSUB, bool RELU, bool OUTBF>
__global__ __launch_bounds__(256) void k_gemm_bf16(const __bf16* __restrict__ A,
                                                   const __bf16* __restrict__ Bt,
                                                   const float* __restrict__ bias,
                                                   float* __restrict__ Cf,
                                                   __bf16* __restrict__ Cbf,
                                                   int N, int K) {
  const int lane = threadIdx.x & 31;
  const int wave = threadIdx.x >> 5;
  const int m0 = blockIdx.x * 16;
  const int n0 = (blockIdx.y * 8 + wave) * 16 * NSUB;
  const int mr = m0 + (lane & 15);
  const int klo = (lane >> 4) << 3;     // lanes 0-15 -> K 0..7/16..23 ; lanes 16-31 -> 8..15/24..31
  const __bf16* arow = A + (size_t)mr * K;
  const __bf16* brow[NSUB];
#pragma unroll
  for (int j = 0; j < NSUB; ++j)
    brow[j] = Bt + (size_t)(n0 + 16 * j + (lane & 15)) * K;

  v8f acc[NSUB];
#pragma unroll
  for (int j = 0; j < NSUB; ++j)
#pragma unroll
    for (int e = 0; e < 8; ++e) acc[j][e] = 0.f;

  for (int k = 0; k < K; k += 32) {
    v8bf a0 = *(const v8bf*)(arow + k + klo);
    v8bf a1 = *(const v8bf*)(arow + k + 16 + klo);
    v16bf av;
#pragma unroll
    for (int e = 0; e < 8; ++e) { av[e] = a0[e]; av[e + 8] = a1[e]; }
#pragma unroll
    for (int j = 0; j < NSUB; ++j) {
      v8bf b0 = *(const v8bf*)(brow[j] + k + klo);
      v8bf b1 = *(const v8bf*)(brow[j] + k + 16 + klo);
      v16bf bv;
#pragma unroll
      for (int e = 0; e < 8; ++e) { bv[e] = b0[e]; bv[e + 8] = b1[e]; }
      acc[j] = __builtin_amdgcn_wmma_f32_16x16x32_bf16(false, av, false, bv, (short)0,
                                                       acc[j], false, false);
    }
  }
  const int rowadd = (lane >> 4) * 8;   // C/D layout: lanes 16-31 hold M = 8..15
#pragma unroll
  for (int j = 0; j < NSUB; ++j) {
    const int nc = n0 + 16 * j + (lane & 15);
    const float bs = bias[nc];
#pragma unroll
    for (int r = 0; r < 8; ++r) {
      int row = m0 + rowadd + r;
      float v = acc[j][r] + bs;
      if (RELU) v = v > 0.f ? v : 0.f;
      if (OUTBF) Cbf[(size_t)row * N + nc] = f2bf(v);
      else       Cf [(size_t)row * N + nc] = v;
    }
  }
}

// ---------------- softmax over 16 (levels*points) per (q, head) ----------------
__global__ void k_softmax16(float* __restrict__ logits) {
  int idx = blockIdx.x * 256 + threadIdx.x;          // (q, head)
  if (idx >= LQ * NHEADS) return;
  float* p = logits + (size_t)idx * 16;
  float mx = p[0];
#pragma unroll
  for (int j = 1; j < 16; ++j) mx = fmaxf(mx, p[j]);
  float s = 0.f;
  float e[16];
#pragma unroll
  for (int j = 0; j < 16; ++j) { e[j] = expf(p[j] - mx); s += e[j]; }
  float inv = 1.f / s;
#pragma unroll
  for (int j = 0; j < 16; ++j) p[j] = e[j] * inv;
}

// ---------------- deformable sampling: wave per (q, head), lane = channel ----------------
static __device__ __forceinline__ float gtap(const float* vb, int yi, int xi, int hl, int wl) {
  if ((unsigned)yi < (unsigned)hl && (unsigned)xi < (unsigned)wl)
    return vb[(size_t)(yi * wl + xi) * CDIM];
  return 0.f;
}

__global__ __launch_bounds__(256) void k_sample(const float* __restrict__ value,   // [LQ, 256]
                                                const float* __restrict__ off,     // [LQ, 256]
                                                const float* __restrict__ aw,      // [LQ, 128] softmaxed
                                                const float* __restrict__ ref,     // [LQ, 2]
                                                __bf16* __restrict__ attn_bf) {    // [LQ, 256]
  const int q = blockIdx.x;
  const int h = threadIdx.x >> 5;
  const int lane = threadIdx.x & 31;
  const float rx = ref[q * 2 + 0];
  const float ry = ref[q * 2 + 1];
  const int hs[4] = {64, 32, 16, 8};
  const int ws[4] = {64, 32, 16, 8};
  const int loff[4] = {0, 4096, 5120, 5376};
  float acc = 0.f;
#pragma unroll
  for (int l = 0; l < NLEV; ++l) {
    const int hl = hs[l], wl = ws[l];
    const float* vb = value + (size_t)loff[l] * CDIM + h * DH + lane;
#pragma unroll
    for (int p = 0; p < NPTS; ++p) {
      int oi = ((h * NLEV + l) * NPTS + p) * 2;
      float lx = rx + off[(size_t)q * CDIM + oi]     / (float)wl;
      float ly = ry + off[(size_t)q * CDIM + oi + 1] / (float)hl;
      float w  = aw[(size_t)q * 128 + (h * NLEV + l) * NPTS + p];
      float x = lx * wl - 0.5f, y = ly * hl - 0.5f;
      float x0f = floorf(x), y0f = floorf(y);
      int x0 = (int)x0f, y0 = (int)y0f;
      float wx = x - x0f, wy = y - y0f;
      float v00 = gtap(vb, y0,     x0,     hl, wl);
      float v01 = gtap(vb, y0,     x0 + 1, hl, wl);
      float v10 = gtap(vb, y0 + 1, x0,     hl, wl);
      float v11 = gtap(vb, y0 + 1, x0 + 1, hl, wl);
      acc += w * ((v00 * (1.f - wx) + v01 * wx) * (1.f - wy) +
                  (v10 * (1.f - wx) + v11 * wx) * wy);
    }
  }
  attn_bf[(size_t)q * CDIM + h * DH + lane] = f2bf(acc);
}

// ---------------- x = LayerNorm(a + b) * g + be ; wave per row ----------------
__global__ void k_add_ln(const float* __restrict__ a, const float* __restrict__ b,
                         const float* __restrict__ g, const float* __restrict__ be,
                         float* __restrict__ X, __bf16* __restrict__ Xbf, int want_bf) {
  int row = blockIdx.x * 8 + (threadIdx.x >> 5);
  int lane = threadIdx.x & 31;
  float v[8];
  float s = 0.f;
#pragma unroll
  for (int i = 0; i < 8; ++i) {
    int c = lane + i * 32;
    v[i] = a[(size_t)row * CDIM + c] + b[(size_t)row * CDIM + c];
    s += v[i];
  }
#pragma unroll
  for (int m = 16; m >= 1; m >>= 1) s += __shfl_xor(s, m, 32);
  float mean = s * (1.f / CDIM);
  float s2 = 0.f;
#pragma unroll
  for (int i = 0; i < 8; ++i) { float d = v[i] - mean; s2 += d * d; }
#pragma unroll
  for (int m = 16; m >= 1; m >>= 1) s2 += __shfl_xor(s2, m, 32);
  float inv = rsqrtf(s2 * (1.f / CDIM) + 1e-5f);
#pragma unroll
  for (int i = 0; i < 8; ++i) {
    int c = lane + i * 32;
    float o = (v[i] - mean) * inv * g[c] + be[c];
    X[(size_t)row * CDIM + c] = o;
    if (want_bf) Xbf[(size_t)row * CDIM + c] = f2bf(o);
  }
}

__global__ void k_copy(const float* __restrict__ a, float* __restrict__ o) {
  size_t i = (size_t)blockIdx.x * 256 + threadIdx.x;
  o[i] = a[i];
}

// =================================================================================
extern "C" void kernel_launch(void* const* d_in, const int* in_sizes, int n_in,
                              void* d_out, int out_size, void* d_ws, size_t ws_size,
                              hipStream_t stream) {
  (void)n_in; (void)out_size; (void)ws_size;

  // ---- map inputs (auto-detect dict-insertion vs alphabetized flattening) ----
  const float *tgt[4], *pos[4], *mem[4], *posm[4];
  const float *le, *W_ref, *b_ref, *W_off, *b_off, *W_attn, *b_attn, *W_val, *b_val;
  const float *W_out, *b_out, *ln1g, *ln1b, *W1, *b1, *W2, *b2, *ln2g, *ln2b;
  bool insertion = in_sizes[0] > 100000;   // tgt0 (1048576) vs mask0 (4096)
  if (insertion) {
    for (int l = 0; l < 4; ++l) {
      tgt[l]  = (const float*)d_in[5 * l + 0];
      pos[l]  = (const float*)d_in[5 * l + 1];
      mem[l]  = (const float*)d_in[5 * l + 2];
      posm[l] = (const float*)d_in[5 * l + 3];
    }
    le = (const float*)d_in[20]; W_ref = (const float*)d_in[21]; b_ref = (const float*)d_in[22];
    W_off = (const float*)d_in[23]; b_off = (const float*)d_in[24];
    W_attn = (const float*)d_in[25]; b_attn = (const float*)d_in[26];
    W_val = (const float*)d_in[27]; b_val = (const float*)d_in[28];
    W_out = (const float*)d_in[29]; b_out = (const float*)d_in[30];
    ln1g = (const float*)d_in[31]; ln1b = (const float*)d_in[32];
    W1 = (const float*)d_in[33]; b1 = (const float*)d_in[34];
    W2 = (const float*)d_in[35]; b2 = (const float*)d_in[36];
    ln2g = (const float*)d_in[37]; ln2b = (const float*)d_in[38];
  } else {
    // alphabetical: mask0-3, mem0-3, params(W1,W2,W_attn,W_off,W_out,W_ref,W_val,
    //               b1,b2,b_attn,b_off,b_out,b_ref,b_val,level_embed,ln1_b,ln1_g,ln2_b,ln2_g),
    //               pos0-3, posm0-3, tgt0-3
    for (int l = 0; l < 4; ++l) {
      mem[l]  = (const float*)d_in[4 + l];
      pos[l]  = (const float*)d_in[27 + l];
      posm[l] = (const float*)d_in[31 + l];
      tgt[l]  = (const float*)d_in[35 + l];
    }
    W1 = (const float*)d_in[8]; W2 = (const float*)d_in[9];
    W_attn = (const float*)d_in[10]; W_off = (const float*)d_in[11];
    W_out = (const float*)d_in[12]; W_ref = (const float*)d_in[13]; W_val = (const float*)d_in[14];
    b1 = (const float*)d_in[15]; b2 = (const float*)d_in[16];
    b_attn = (const float*)d_in[17]; b_off = (const float*)d_in[18]; b_out = (const float*)d_in[19];
    b_ref = (const float*)d_in[20]; b_val = (const float*)d_in[21];
    le = (const float*)d_in[22];
    ln1b = (const float*)d_in[23]; ln1g = (const float*)d_in[24];
    ln2b = (const float*)d_in[25]; ln2g = (const float*)d_in[26];
  }

  // ---- workspace carve ----
  char* wp = (char*)d_ws;
  auto alloc = [&](size_t bytes) -> void* {
    void* p = (void*)wp;
    wp += (bytes + 255) & ~(size_t)255;
    return p;
  };
  const size_t QC = (size_t)LQ * CDIM;
  float*  vin    = (float*) alloc(QC * 4);
  __bf16* vinbf  = (__bf16*)alloc(QC * 2);
  float*  pe     = (float*) alloc(QC * 4);
  float*  posraw = (float*) alloc(QC * 4);
  float*  refbuf = (float*) alloc((size_t)LQ * 2 * 4);
  float*  outb   = (float*) alloc(QC * 4);
  __bf16* qbf    = (__bf16*)alloc(QC * 2);
  float*  value  = (float*) alloc(QC * 4);
  float*  offb   = (float*) alloc(QC * 4);
  float*  logits = (float*) alloc((size_t)LQ * 128 * 4);
  __bf16* attnbf = (__bf16*)alloc(QC * 2);
  float*  src2   = (float*) alloc(QC * 4);
  float*  xbuf   = (float*) alloc(QC * 4);
  __bf16* xbf    = (__bf16*)alloc(QC * 2);
  __bf16* hbf    = (__bf16*)alloc((size_t)LQ * DFFN * 2);
  float*  ffn    = (float*) alloc(QC * 4);
  __bf16* Wval_t  = (__bf16*)alloc((size_t)NLAYERS * CDIM * CDIM * 2);
  __bf16* Woff_t  = (__bf16*)alloc((size_t)NLAYERS * CDIM * CDIM * 2);
  __bf16* Wattn_t = (__bf16*)alloc((size_t)NLAYERS * CDIM * 128 * 2);
  __bf16* Wout_t  = (__bf16*)alloc((size_t)NLAYERS * CDIM * CDIM * 2);
  __bf16* W1_t    = (__bf16*)alloc((size_t)NLAYERS * CDIM * DFFN * 2);
  __bf16* W2_t    = (__bf16*)alloc((size_t)NLAYERS * DFFN * CDIM * 2);

  // ---- weight transpose/cast (once per launch) ----
  {
    dim3 g256((CDIM * CDIM + 255) / 256, NLAYERS);
    k_wt_cast<<<g256, 256, 0, stream>>>(W_val, Wval_t, CDIM, CDIM);
    k_wt_cast<<<g256, 256, 0, stream>>>(W_off, Woff_t, CDIM, CDIM);
    k_wt_cast<<<g256, 256, 0, stream>>>(W_out, Wout_t, CDIM, CDIM);
    dim3 g128((CDIM * 128 + 255) / 256, NLAYERS);
    k_wt_cast<<<g128, 256, 0, stream>>>(W_attn, Wattn_t, CDIM, 128);
    dim3 gff((CDIM * DFFN + 255) / 256, NLAYERS);
    k_wt_cast<<<gff, 256, 0, stream>>>(W1, W1_t, CDIM, DFFN);
    k_wt_cast<<<gff, 256, 0, stream>>>(W2, W2_t, DFFN, CDIM);
  }

  // ---- prep: concat levels, level embed, value_in, ref points ----
  {
    const int ns[4]   = {4096, 1024, 256, 64};
    const int qoff[4] = {0, 4096, 5120, 5376};
    for (int l = 0; l < 4; ++l) {
      k_prep_level<<<ns[l], 256, 0, stream>>>(tgt[l], pos[l], mem[l], posm[l], le + l * CDIM,
                                              pe, posraw, outb, vin, vinbf, ns[l], qoff[l]);
    }
    k_ref<<<LQ / 8, 256, 0, stream>>>(posraw, W_ref, b_ref, refbuf);
  }

  // ---- layers ----
  const int MT = LQ / 16;  // 340
  for (int i = 0; i < NLAYERS; ++i) {
    // q = out + pos_embed (bf16)
    k_qcast<<<(int)(QC / 256), 256, 0, stream>>>(outb, pe, qbf);
    // value = value_in @ W_val + b_val            [N=256, NSUB=2 -> grid (340,1)]
    k_gemm_bf16<2, false, false><<<dim3(MT, 1), 256, 0, stream>>>(
        vinbf, Wval_t + (size_t)i * CDIM * CDIM, b_val + i * CDIM, value, nullptr, CDIM, CDIM);
    // off = q @ W_off + b_off
    k_gemm_bf16<2, false, false><<<dim3(MT, 1), 256, 0, stream>>>(
        qbf, Woff_t + (size_t)i * CDIM * CDIM, b_off + i * CDIM, offb, nullptr, CDIM, CDIM);
    // attn logits = q @ W_attn + b_attn           [N=128, NSUB=1]
    k_gemm_bf16<1, false, false><<<dim3(MT, 1), 256, 0, stream>>>(
        qbf, Wattn_t + (size_t)i * CDIM * 128, b_attn + i * 128, logits, nullptr, 128, CDIM);
    k_softmax16<<<(LQ * NHEADS + 255) / 256, 256, 0, stream>>>(logits);
    // deformable bilinear sampling -> attn_out (bf16)
    k_sample<<<LQ, 256, 0, stream>>>(value, offb, logits, refbuf, attnbf);
    // src2 = attn_out @ W_out + b_out
    k_gemm_bf16<2, false, false><<<dim3(MT, 1), 256, 0, stream>>>(
        attnbf, Wout_t + (size_t)i * CDIM * CDIM, b_out + i * CDIM, src2, nullptr, CDIM, CDIM);
    // x = LN(out + src2)
    k_add_ln<<<LQ / 8, 256, 0, stream>>>(outb, src2, ln1g + i * CDIM, ln1b + i * CDIM,
                                         xbuf, xbf, 1);
    // h = relu(x @ W1 + b1) (bf16 epilogue)       [N=1024, NSUB=4 -> grid (340,2)]
    k_gemm_bf16<4, true, true><<<dim3(MT, 2), 256, 0, stream>>>(
        xbf, W1_t + (size_t)i * CDIM * DFFN, b1 + i * DFFN, nullptr, hbf, DFFN, CDIM);
    // ffn = h @ W2 + b2                           [N=256, K=1024, NSUB=2]
    k_gemm_bf16<2, false, false><<<dim3(MT, 1), 256, 0, stream>>>(
        hbf, W2_t + (size_t)i * DFFN * CDIM, b2 + i * CDIM, ffn, nullptr, CDIM, DFFN);
    // out = LN(x + ffn)
    k_add_ln<<<LQ / 8, 256, 0, stream>>>(xbuf, ffn, ln2g + i * CDIM, ln2b + i * CDIM,
                                         outb, nullptr, 0);
  }

  // ---- final output (fp32) ----
  k_copy<<<(int)(QC / 256), 256, 0, stream>>>(outb, (float*)d_out);
}
